// OptimizedLinearCrossEntropy_52965536694582
// MI455X (gfx1250) — matrix operand order repaired and checked by reading, not compiled, not run on hardware
//
#include <hip/hip_runtime.h>
#include <cstdint>

#define BT 2048
#define HD 2048
#define VV 50257
#define NCHUNK 197
#define TILES_PER_CHUNK 16          // 197*16 = 3152 col-tiles >= ceil(50257/16)=3142
#define NPART (NCHUNK * 8)          // per-(chunk,wave) partials
#define KSTEPS (HD / 32)            // 64 WMMA K-steps of 32

typedef __attribute__((ext_vector_type(16))) __bf16 v16bf;
typedef __attribute__((ext_vector_type(8)))  float  v8f;
typedef __attribute__((ext_vector_type(4)))  unsigned int v4u;
typedef __attribute__((ext_vector_type(8)))  int v8i;
typedef __attribute__((ext_vector_type(4)))  int v4i;

union ABfrag { v16bf v; uint4 q[2]; };

#if __has_builtin(__builtin_amdgcn_tensor_load_to_lds)
#define HAVE_TDM 1
#else
#define HAVE_TDM 0
#endif

__device__ __forceinline__ unsigned short f2bf(float f) {
    unsigned u = __float_as_uint(f);
    u += 0x7FFFu + ((u >> 16) & 1u);          // round-to-nearest-even
    return (unsigned short)(u >> 16);
}

// ---------------------------------------------------------------- convert f32 -> bf16
__global__ __launch_bounds__(256) void cvt_f32_bf16(const float* __restrict__ in,
                                                    unsigned short* __restrict__ out,
                                                    int n4) {
    int i = blockIdx.x * 256 + threadIdx.x;
    if (i >= n4) return;
    float4 f = ((const float4*)in)[i];
    ushort4 o;
    o.x = f2bf(f.x); o.y = f2bf(f.y); o.z = f2bf(f.z); o.w = f2bf(f.w);
    ((ushort4*)out)[i] = o;
}

__global__ void zero_accum(float* a) {
    if (threadIdx.x < 2) a[threadIdx.x] = 0.0f;
}

#if HAVE_TDM
// Tensor Data Mover: stage one 16 x 2048 bf16 row-tile of x into LDS.
// D# per CDNA5 ISA: group0 = count/lds_addr/global_addr/type,
// group1 = data_size + dims/strides/tile dims. 2D tile -> groups 2/3 zero.
__device__ __forceinline__ void tdm_load_tile(const unsigned short* xB, int rt,
                                              unsigned short* dst) {
    unsigned long long ga = (unsigned long long)(uintptr_t)(xB + (size_t)rt * 16 * HD);
    unsigned int ldsa = (unsigned int)(uintptr_t)dst;   // low 32 bits = LDS offset
    v4u g0;
    g0[0] = 0x1u;                                        // count=1, is_restore=0
    g0[1] = ldsa;                                        // lds_addr  (bits 63:32)
    g0[2] = (unsigned int)ga;                            // global_addr[31:0] (bits 95:64)
    g0[3] = (unsigned int)((ga >> 32) & 0x01FFFFFFull)   // global_addr[56:32]
          | (2u << 30);                                  // type = 2 ("image")
    v8i g1;
    g1[0] = 0x00010000;                                  // data_size=1 -> 2-byte elems
    g1[1] = (int)(HD << 16);                             // tensor_dim0[15:0] @ bits 63:48
    g1[2] = (int)(BT << 16);                             // tensor_dim1[15:0] @ bits 95:80
    g1[3] = (int)(HD << 16);                             // tile_dim0 @ bits 127:112
    g1[4] = 16;                                          // tile_dim1 = 16 rows, tile_dim2 = 0
    g1[5] = HD;                                          // tensor_dim0_stride[31:0]
    g1[6] = 0;                                           // stride hi / dim1_stride lo
    g1[7] = 0;
    v4i g2 = {0, 0, 0, 0};
    v4i g3 = {0, 0, 0, 0};
#if defined(__clang_major__) && (__clang_major__ >= 23)
    v8i g4 = {0, 0, 0, 0, 0, 0, 0, 0};
    __builtin_amdgcn_tensor_load_to_lds(g0, g1, g2, g3, g4, 0);
#else
    __builtin_amdgcn_tensor_load_to_lds(g0, g1, g2, g3, 0);
#endif
}
#endif

// ---------------------------------------------------------------- main fused GEMM + online softmax stats
// grid.x = NCHUNK column chunks; 256 threads = 8 waves. Each wave owns col-tiles
// (wave, wave+8) of the chunk, sharing A fragments across both -> 2 WMMAs per A load.
// Block loops over all 128 row-tiles; its ~1MB weight slice stays L2-resident.
// TDM double-buffers the x row-tile (2 x 64KB LDS) one iteration ahead.
__global__ __launch_bounds__(256) void lce_main(const unsigned short* __restrict__ xB,
                                                const unsigned short* __restrict__ wB,
                                                const float* __restrict__ bias,
                                                float* __restrict__ part_max,
                                                float* __restrict__ part_sum) {
    __shared__ unsigned short lds[2][16 * HD];   // 2 x 64KB double buffer

    const int chunk = blockIdx.x;
    const int wave  = threadIdx.x >> 5;       // 0..7
    const int lane  = threadIdx.x & 31;
    const int half  = lane >> 4;              // 0 | 1
    const int lr    = lane & 15;

    // this wave's two column tiles (wave-uniform bases)
    const int n0base = (chunk * TILES_PER_CHUNK + wave) * 16;
    const int n1base = n0base + 8 * 16;
    const bool anyv  = (n0base < VV);         // n1base < VV implies n0base < VV
    const int  col0  = n0base + lr;
    const int  col1  = n1base + lr;
    const bool c0v   = (col0 < VV);
    const bool c1v   = (col1 < VV);
    const unsigned short* wrow0 = wB + (size_t)(c0v ? col0 : 0) * HD;
    const unsigned short* wrow1 = wB + (size_t)(c1v ? col1 : 0) * HD;

#if HAVE_TDM
    if (wave == 0) tdm_load_tile(xB, 0, &lds[0][0]);
#endif

    for (int rt = 0; rt < BT / 16; ++rt) {
        const int buf = rt & 1;
#if HAVE_TDM
        if (wave == 0) __builtin_amdgcn_s_wait_tensorcnt(0);
        __syncthreads();                       // lds[buf] ready; lds[buf^1] reads done
        if (wave == 0 && rt + 1 < BT / 16) tdm_load_tile(xB, rt + 1, &lds[buf ^ 1][0]);
#else
        __syncthreads();
        for (int c = threadIdx.x; c < 16 * 256; c += 256) {
            int row = c >> 8, k8 = c & 255;
            *(uint4*)&lds[buf][row * HD + k8 * 8] =
                *(const uint4*)(xB + (size_t)(rt * 16 + row) * HD + k8 * 8);
        }
        __syncthreads();
#endif
        const unsigned short* xa = &lds[buf][0];

        float run_max[8], run_sum[8];
#pragma unroll
        for (int i = 0; i < 8; ++i) { run_max[i] = -INFINITY; run_sum[i] = 0.0f; }

        if (anyv) {
            v8f acc00 = {}, acc01 = {}, acc10 = {}, acc11 = {};
#pragma unroll 2
            for (int kt = 0; kt < KSTEPS; kt += 2) {
                // A fragments, addresses linear in kt (fold into ds_load offsets)
                ABfrag af0, af1;
                const unsigned short* ap = xa + lr * HD + kt * 32 + half * 16;
                af0.q[0] = *(const uint4*)(ap);
                af0.q[1] = *(const uint4*)(ap + 8);
                af1.q[0] = *(const uint4*)(ap + 32);
                af1.q[1] = *(const uint4*)(ap + 40);
                // B fragments for both tiles, both k-steps (32B contiguous per lane)
                ABfrag b00, b01, b10, b11;
                const uint4* bp0 = (const uint4*)(wrow0 + kt * 32 + half * 16);
                const uint4* bp1 = (const uint4*)(wrow1 + kt * 32 + half * 16);
                b00.q[0] = bp0[0]; b00.q[1] = bp0[1];
                b01.q[0] = bp0[2]; b01.q[1] = bp0[3];
                b10.q[0] = bp1[0]; b10.q[1] = bp1[1];
                b11.q[0] = bp1[2]; b11.q[1] = bp1[3];

                acc00 = __builtin_amdgcn_wmma_f32_16x16x32_bf16(
                    false, af0.v, false, b00.v, (short)0, acc00, false, false);
                acc10 = __builtin_amdgcn_wmma_f32_16x16x32_bf16(
                    false, af0.v, false, b10.v, (short)0, acc10, false, false);
                acc01 = __builtin_amdgcn_wmma_f32_16x16x32_bf16(
                    false, af1.v, false, b01.v, (short)0, acc01, false, false);
                acc11 = __builtin_amdgcn_wmma_f32_16x16x32_bf16(
                    false, af1.v, false, b11.v, (short)0, acc11, false, false);
            }

            float bv0 = c0v ? bias[col0] : 0.0f;
            float bv1 = c1v ? bias[col1] : 0.0f;
#pragma unroll
            for (int i = 0; i < 8; ++i) {
                // ---- tile 0 contribution
                float val = c0v ? (acc00[i] + acc01[i] + bv0) : -INFINITY;
                float m = val;
#pragma unroll
                for (int d = 1; d < 16; d <<= 1) m = fmaxf(m, __shfl_xor(m, d, 16));
                float nm = fmaxf(run_max[i], m);
                float e = (val > -INFINITY) ? __expf(val - nm) : 0.0f;
#pragma unroll
                for (int d = 1; d < 16; d <<= 1) e += __shfl_xor(e, d, 16);
                float scale = (run_max[i] > -INFINITY) ? __expf(run_max[i] - nm) : 0.0f;
                run_sum[i] = run_sum[i] * scale + e;
                run_max[i] = nm;
                // ---- tile 1 contribution
                val = c1v ? (acc10[i] + acc11[i] + bv1) : -INFINITY;
                m = val;
#pragma unroll
                for (int d = 1; d < 16; d <<= 1) m = fmaxf(m, __shfl_xor(m, d, 16));
                nm = fmaxf(run_max[i], m);
                e = (val > -INFINITY) ? __expf(val - nm) : 0.0f;
#pragma unroll
                for (int d = 1; d < 16; d <<= 1) e += __shfl_xor(e, d, 16);
                scale = (run_max[i] > -INFINITY) ? __expf(run_max[i] - nm) : 0.0f;
                run_sum[i] = run_sum[i] * scale + e;
                run_max[i] = nm;
            }
        }

        if (lr == 0) {  // lanes 0 and 16 own rows [0..7]+8*half of this tile
            int base = (chunk * 8 + wave) * BT + rt * 16 + half * 8;
#pragma unroll
            for (int i = 0; i < 8; ++i) {
                part_max[base + i] = run_max[i];
                part_sum[base + i] = run_sum[i];
            }
        }
    }
}

// ---------------------------------------------------------------- per-row: merge partials, exact f32 target logit, NLL
__global__ __launch_bounds__(256) void lce_row(const float* __restrict__ x,
                                               const float* __restrict__ w,
                                               const float* __restrict__ bias,
                                               const int* __restrict__ target,
                                               const float* __restrict__ part_max,
                                               const float* __restrict__ part_sum,
                                               float* __restrict__ accum) {
    __shared__ float red[256];
    const int row = blockIdx.x;
    const int tid = threadIdx.x;

    int  tgt   = target[row];
    bool valid = (tgt != -100);
    int  tsafe = (valid && tgt >= 0 && tgt < VV) ? tgt : 0;

    // exact fp32 dot for the target logit
    const float* xr = x + (size_t)row * HD;
    const float* wr = w + (size_t)tsafe * HD;
    float p = 0.0f;
    for (int k = tid; k < HD; k += 256) p = fmaf(xr[k], wr[k], p);
    red[tid] = p; __syncthreads();
    for (int s = 128; s > 0; s >>= 1) { if (tid < s) red[tid] += red[tid + s]; __syncthreads(); }
    float logit_t = red[0] + bias[tsafe];
    __syncthreads();

    // global max over NPART partials
    float m = -INFINITY;
    for (int j = tid; j < NPART; j += 256) m = fmaxf(m, part_max[(size_t)j * BT + row]);
    red[tid] = m; __syncthreads();
    for (int s = 128; s > 0; s >>= 1) { if (tid < s) red[tid] = fmaxf(red[tid], red[tid + s]); __syncthreads(); }
    float M = red[0];
    __syncthreads();

    // merged sum-exp
    float ssum = 0.0f;
    for (int j = tid; j < NPART; j += 256) {
        float pm = part_max[(size_t)j * BT + row];
        float ps = part_sum[(size_t)j * BT + row];
        ssum += (pm > -INFINITY) ? ps * __expf(pm - M) : 0.0f;
    }
    red[tid] = ssum; __syncthreads();
    for (int s = 128; s > 0; s >>= 1) { if (tid < s) red[tid] += red[tid + s]; __syncthreads(); }

    if (tid == 0) {
        float lse  = M + __logf(red[0]);
        float loss = valid ? (lse - logit_t) : 0.0f;
        atomicAdd(&accum[0], loss);
        atomicAdd(&accum[1], valid ? 1.0f : 0.0f);
    }
}

__global__ void finalize(const float* accum, float* out) {
    out[0] = accum[0] / fmaxf(accum[1], 1.0f);
}

// ----------------------------------------------------------------
extern "C" void kernel_launch(void* const* d_in, const int* in_sizes, int n_in,
                              void* d_out, int out_size, void* d_ws, size_t ws_size,
                              hipStream_t stream) {
    const float* x      = (const float*)d_in[0];
    const int*   target = (const int*)d_in[1];
    const float* weight = (const float*)d_in[2];
    const float* bias   = (const float*)d_in[3];
    float*       out    = (float*)d_out;

    // workspace layout
    char* ws = (char*)d_ws;
    unsigned short* wB = (unsigned short*)ws;               // 50257*2048*2  ~206 MB
    ws += (size_t)VV * HD * 2;
    unsigned short* xB = (unsigned short*)ws;               // 2048*2048*2   8 MB
    ws += (size_t)BT * HD * 2;
    float* part_max = (float*)ws;                            // NPART*BT*4   ~12.9 MB
    ws += (size_t)NPART * BT * 4;
    float* part_sum = (float*)ws;                            // NPART*BT*4   ~12.9 MB
    ws += (size_t)NPART * BT * 4;
    float* accum = (float*)ws;                               // 2 floats

    int nW4 = (VV * HD) / 4;
    int nX4 = (BT * HD) / 4;
    cvt_f32_bf16<<<(nW4 + 255) / 256, 256, 0, stream>>>(weight, wB, nW4);
    cvt_f32_bf16<<<(nX4 + 255) / 256, 256, 0, stream>>>(x, xB, nX4);
    zero_accum<<<1, 32, 0, stream>>>(accum);
    lce_main<<<NCHUNK, 256, 0, stream>>>(xB, wB, bias, part_max, part_sum);
    lce_row<<<BT, 256, 0, stream>>>(x, weight, bias, target, part_max, part_sum, accum);
    finalize<<<1, 1, 0, stream>>>(accum, out);
}
